// MultiHeadAttention_54142357733640
// MI455X (gfx1250) — matrix operand order repaired and checked by reading, not compiled, not run on hardware
//
#include <hip/hip_runtime.h>
#include <hip/hip_bf16.h>

// MHA forward for B=2, S=2048, D=1024, H=16, DK=DV=64 on gfx1250 (wave32, WMMA).
// Pipeline: f32->bf16 convert (vectorized), 3x projection GEMM (bf16 WMMA,
// TDM-staged A tiles), flash-attention (TDM-staged K, bf16 WMMA, online
// softmax), output GEMM (f32 out).
// All LDS fragment traffic is 16B-vectorized (ds_load_b128-friendly layouts).

#define BATCH 2
#define SEQ   2048
#define DMODEL 1024
#define NH    16
#define DHEAD 64
#define MROWS (BATCH * SEQ)   // 4096

typedef __attribute__((ext_vector_type(16))) __bf16          v16bf;
typedef __attribute__((ext_vector_type(16))) unsigned short  v16u;
typedef __attribute__((ext_vector_type(8)))  float           v8f;
typedef __attribute__((ext_vector_type(4)))  unsigned int    v4u;
typedef __attribute__((ext_vector_type(8)))  int             v8i;
typedef __attribute__((ext_vector_type(4)))  int             v4i;

static __device__ __forceinline__ unsigned short f32_to_bf16(float f) {
  union { float f; unsigned int u; } v; v.f = f;
  unsigned int lsb = (v.u >> 16) & 1u;
  unsigned int r = v.u + 0x7fffu + lsb;   // round-to-nearest-even
  return (unsigned short)(r >> 16);
}

static __device__ __forceinline__ v8f wmma_bf16(v16u a, v16u b, v8f c) {
  return __builtin_amdgcn_wmma_f32_16x16x32_bf16(
      /*neg_a=*/false, __builtin_bit_cast(v16bf, a),
      /*neg_b=*/false, __builtin_bit_cast(v16bf, b),
      /*c_mod=*/(short)0, c, /*reuse_a=*/false, /*reuse_b=*/false);
}

// Build a 16-element bf16 fragment from two contiguous 16B runs (b128 loads).
struct FragPair { v4u lo; v4u hi; };
static __device__ __forceinline__ v16u make_frag(const unsigned short* p0,
                                                 const unsigned short* p1) {
  FragPair f;
  f.lo = *reinterpret_cast<const v4u*>(p0);
  f.hi = *reinterpret_cast<const v4u*>(p1);
  return __builtin_bit_cast(v16u, f);
}

// Issue a TDM load of a 2-D bf16 tile (rows x cols) into LDS with 4-dword
// padding every 32 dwords (matches a 72-ushort LDS pitch for 64-elem rows).
// D# layout per CDNA5 ISA.
static __device__ __forceinline__ void tdm_load_tile_2d(
    unsigned lds_addr, unsigned long long gaddr,
    int tile_cols, int tile_rows, int tensor_d0, int tensor_d1,
    int row_stride_elems, unsigned pad_bits) {
  v4u g0;
  g0[0] = 1u;                                   // count=1, user descriptor
  g0[1] = lds_addr;                             // lds_addr (bytes)
  g0[2] = (unsigned)(gaddr & 0xffffffffull);    // global_addr[31:0]
  g0[3] = (unsigned)((gaddr >> 32) & 0x1ffffffull) | (2u << 30); // type=2
  v8i g1;
  g1[0] = (int)((1u << 16) | pad_bits);         // data_size=2B (+ padding cfg)
  g1[1] = (int)(((unsigned)tensor_d0 & 0xffffu) << 16);  // tensor_dim0[15:0]
  g1[2] = (int)((((unsigned)tensor_d0 >> 16) & 0xffffu) |
                (((unsigned)tensor_d1 & 0xffffu) << 16)); // d0 hi | d1 lo
  g1[3] = (int)((((unsigned)tensor_d1 >> 16) & 0xffffu) |
                ((unsigned)tile_cols << 16));   // d1 hi | tile_dim0
  g1[4] = tile_rows;                            // tile_dim1 (tile_dim2=0)
  g1[5] = row_stride_elems;                     // tensor_dim0_stride[31:0]
  g1[6] = 0;
  g1[7] = 0;
  v4i g2; g2[0] = 1; g2[1] = 1; g2[2] = 0; g2[3] = 0;  // dim2/3 = 1
  v4i g3; g3[0] = 0; g3[1] = 0; g3[2] = 0; g3[3] = 0;
  v8i g4; g4[0] = 0; g4[1] = 0; g4[2] = 0; g4[3] = 0;
  g4[4] = 0; g4[5] = 0; g4[6] = 0; g4[7] = 0;
  __builtin_amdgcn_tensor_load_to_lds(g0, g1, g2, g3, g4, 0);
}

// pad_enable | pad_interval=32dw | pad_amount=4dw  (64 bf16 rows -> pitch 72)
#define PAD_32DW_4DW ((1u << 20) | (4u << 22) | (3u << 25))

// ---------------------------------------------------------------------------
// fp32 -> bf16 elementwise convert (4 elems/thread/iter).
// ---------------------------------------------------------------------------
__global__ void cvt_f32_bf16(const float* __restrict__ src,
                             unsigned short* __restrict__ dst, int n4) {
  int i = blockIdx.x * blockDim.x + threadIdx.x;
  int stride = gridDim.x * blockDim.x;
  for (; i < n4; i += stride) {
    float4 f = reinterpret_cast<const float4*>(src)[i];
    uint2 p;
    p.x = (unsigned)f32_to_bf16(f.x) | ((unsigned)f32_to_bf16(f.y) << 16);
    p.y = (unsigned)f32_to_bf16(f.z) | ((unsigned)f32_to_bf16(f.w) << 16);
    reinterpret_cast<uint2*>(dst)[i] = p;
  }
}

// ---------------------------------------------------------------------------
// Tiled bf16 GEMM: C[M,N] = A[M,K] * B[K,N] + bias, out bf16 or f32.
// 128x128 block tile, K-step 64. 256 threads = 8 waves (4x2); each wave does
// a 32x64 patch = 2x4 WMMA f32_16x16x32_bf16 accumulators, 2 K-halves/stage.
// A tile staged by the Tensor Data Mover; B staged K-major (transposed) so
// every fragment is two contiguous 16B runs -> ds_load_b128.
// ---------------------------------------------------------------------------
#define TM 128
#define TN 128
#define TK 64
#define APITCH 72   // 64 data + 8 pad ushorts; 144B = 36 dwords
#define BPITCH 72

__global__ __launch_bounds__(256) void gemm_bf16(
    const unsigned short* __restrict__ A,
    const unsigned short* __restrict__ Bm,
    const float* __restrict__ bias,
    unsigned short* __restrict__ outB,
    float* __restrict__ outF,
    int M, int N, int K, int out_is_f32) {
  __shared__ __align__(16) unsigned short As[TM][APITCH];   // row-major [m][k]
  __shared__ __align__(16) unsigned short Bsc[TN][BPITCH];  // col-major [n][k]

  const int tid  = threadIdx.x;
  const int lane = tid & 31;
  const int wave = tid >> 5;
  const int wm   = wave >> 1;      // 0..3
  const int wn   = wave & 1;       // 0..1
  const int ll   = lane & 15;
  const int lh   = lane >> 4;
  const int m0   = blockIdx.y * TM;
  const int n0   = blockIdx.x * TN;
  const unsigned ldsAs = (unsigned)(unsigned long long)(uintptr_t)&As[0][0];

  v8f acc[2][4];
#pragma unroll
  for (int i = 0; i < 2; ++i)
#pragma unroll
    for (int j = 0; j < 4; ++j) {
      v8f z = {0.f, 0.f, 0.f, 0.f, 0.f, 0.f, 0.f, 0.f};
      acc[i][j] = z;
    }

  for (int k0 = 0; k0 < K; k0 += TK) {
    __syncthreads();   // previous iteration's LDS reads complete
    // A tile 128x64 via TDM (DMA straight into padded LDS rows).
    if (wave == 0) {
      unsigned long long gaddr =
          (unsigned long long)(uintptr_t)(A + (size_t)m0 * K + k0);
      tdm_load_tile_2d(ldsAs, gaddr, /*cols*/TK, /*rows*/TM,
                       /*tensor_d0*/K, /*tensor_d1*/M,
                       /*stride*/K, PAD_32DW_4DW);
    }
    // Stage B tile transposed: read row chunks, scatter into column-major LDS.
#pragma unroll
    for (int g = tid; g < (TK * TN) / 4; g += 256) {
      int krow = g >> 5;
      int c4   = (g & 31) << 2;
      uint2 d = *reinterpret_cast<const uint2*>(
          &Bm[(size_t)(k0 + krow) * N + n0 + c4]);
      Bsc[c4 + 0][krow] = (unsigned short)(d.x & 0xffffu);
      Bsc[c4 + 1][krow] = (unsigned short)(d.x >> 16);
      Bsc[c4 + 2][krow] = (unsigned short)(d.y & 0xffffu);
      Bsc[c4 + 3][krow] = (unsigned short)(d.y >> 16);
    }
    if (wave == 0) __builtin_amdgcn_s_wait_tensorcnt(0);
    __syncthreads();

    // Prefetch next K-tile of B into L2 (global_prefetch_b8).
    if (k0 + TK < K)
      __builtin_prefetch(&Bm[(size_t)(k0 + TK + (tid >> 3)) * N + n0 + ((tid & 7) << 4)], 0, 1);

#pragma unroll
    for (int kh = 0; kh < 2; ++kh) {
      // A fragments: lane ll = row, K run kh*32+lh*16..+15 (contiguous).
      v16u af[2];
#pragma unroll
      for (int mt = 0; mt < 2; ++mt) {
        const unsigned short* ar = &As[wm * 32 + mt * 16 + ll][kh * 32];
        af[mt] = make_frag(ar + (lh << 4), ar + (lh << 4) + 8);
      }
      // B fragments: lane ll = col, K run kh*32+lh*16..+15 (contiguous).
      v16u bfrag[4];
#pragma unroll
      for (int nt = 0; nt < 4; ++nt) {
        const unsigned short* br = &Bsc[wn * 64 + nt * 16 + ll][kh * 32 + (lh << 4)];
        bfrag[nt] = make_frag(br, br + 8);
      }
#pragma unroll
      for (int mt = 0; mt < 2; ++mt)
#pragma unroll
        for (int nt = 0; nt < 4; ++nt)
          acc[mt][nt] = wmma_bf16(af[mt], bfrag[nt], acc[mt][nt]);
    }
  }

  // Epilogue: bias add, store. C layout: row = r + 8*lh, col = ll in tile.
#pragma unroll
  for (int mt = 0; mt < 2; ++mt)
#pragma unroll
    for (int nt = 0; nt < 4; ++nt) {
      int col = n0 + wn * 64 + nt * 16 + ll;
      float bval = bias[col];
#pragma unroll
      for (int r = 0; r < 8; ++r) {
        int row = m0 + wm * 32 + mt * 16 + r + lh * 8;
        float val = acc[mt][nt][r] + bval;
        if (out_is_f32) outF[(size_t)row * N + col] = val;
        else            outB[(size_t)row * N + col] = f32_to_bf16(val);
      }
    }
}

// ---------------------------------------------------------------------------
// Flash attention: grid (S/64, H, B), 128 threads = 4 waves.
// K blocks staged by the Tensor Data Mover with LDS padding so b128 fragment
// loads line up; V staged transposed manually (TDM cannot transpose).
// ---------------------------------------------------------------------------
#define KPITCH 72   // 64 + 8 ushorts
#define VPITCH 40   // Vst row stride (col-major V)

__global__ __launch_bounds__(128) void attn_kernel(
    const unsigned short* __restrict__ qp,
    const unsigned short* __restrict__ kp,
    const unsigned short* __restrict__ vp,
    const unsigned char* __restrict__ mask,
    unsigned short* __restrict__ ctx) {
  __shared__ __align__(16) unsigned short Ks[32][KPITCH];    // row-major K
  __shared__ __align__(16) unsigned short Vst[DHEAD][VPITCH];// col-major V
  __shared__ __align__(16) unsigned short Ps[4][16][32];     // per-wave P

  const int tid  = threadIdx.x;
  const int lane = tid & 31;
  const int wave = tid >> 5;
  const int ll   = lane & 15;
  const int lh   = lane >> 4;
  const int h    = blockIdx.y;
  const int b    = blockIdx.z;
  const int q0   = blockIdx.x * 64 + wave * 16;
  const size_t rowstride = (size_t)NH * DHEAD;                  // 1024
  const size_t base = ((size_t)b * SEQ) * rowstride + (size_t)h * DHEAD;
  const unsigned ldsKs = (unsigned)(unsigned long long)(uintptr_t)&Ks[0][0];

  // q tile: two 16x32 A fragments straight from global (b128 loads).
  v16u aq[2];
  {
    const unsigned short* qrow = qp + base + (size_t)(q0 + ll) * rowstride;
#pragma unroll
    for (int f = 0; f < 2; ++f)
      aq[f] = make_frag(qrow + f * 32 + (lh << 3),
                        qrow + f * 32 + 16 + (lh << 3));
  }

  // Per-lane running mask pointer: row q0+r+8*lh, col kv0+nt*16+ll.
  const unsigned char* mp =
      mask + ((size_t)b * SEQ + q0 + lh * 8) * SEQ + ll;

  float mrow[8], lrow[8];
  v8f o[4];
#pragma unroll
  for (int r = 0; r < 8; ++r) { mrow[r] = -3.0e38f; lrow[r] = 0.f; }
#pragma unroll
  for (int vt = 0; vt < 4; ++vt) {
    v8f z = {0.f, 0.f, 0.f, 0.f, 0.f, 0.f, 0.f, 0.f};
    o[vt] = z;
  }
  const float scale = 0.125f;   // 1/sqrt(64)

  for (int kv0 = 0; kv0 < SEQ; kv0 += 32) {
    __syncthreads();

    // K block 32x64 via TDM.
    if (wave == 0) {
      unsigned long long gaddr =
          (unsigned long long)(uintptr_t)(kp + base + (size_t)kv0 * rowstride);
      tdm_load_tile_2d(ldsKs, gaddr, /*cols*/DHEAD, /*rows*/32,
                       /*tensor_d0*/DMODEL, /*tensor_d1*/MROWS,
                       /*stride*/DMODEL, PAD_32DW_4DW);
    }

    // Stage V transposed: 32x64 -> Vst[dv][kv] (scatter b16 stores).
#pragma unroll
    for (int g = tid; g < (32 * DHEAD) / 4; g += 128) {
      int krow = g >> 4;
      int c4   = (g & 15) << 2;
      uint2 d = *reinterpret_cast<const uint2*>(
          &vp[base + (size_t)(kv0 + krow) * rowstride + c4]);
      Vst[c4 + 0][krow] = (unsigned short)(d.x & 0xffffu);
      Vst[c4 + 1][krow] = (unsigned short)(d.x >> 16);
      Vst[c4 + 2][krow] = (unsigned short)(d.y & 0xffffu);
      Vst[c4 + 3][krow] = (unsigned short)(d.y >> 16);
    }

    if (wave == 0) __builtin_amdgcn_s_wait_tensorcnt(0);
    __syncthreads();

    // Scores: q (16x64) . K^T -> two 16x16 tiles; dk contiguous in Ks rows.
    v8f sc[2];
#pragma unroll
    for (int nt = 0; nt < 2; ++nt) {
      v8f s = {0.f, 0.f, 0.f, 0.f, 0.f, 0.f, 0.f, 0.f};
#pragma unroll
      for (int f = 0; f < 2; ++f) {
        const unsigned short* kr = &Ks[nt * 16 + ll][f * 32];
        s = wmma_bf16(aq[f], make_frag(kr + (lh << 3), kr + 16 + (lh << 3)), s);
      }
      sc[nt] = s;
    }

    // Scale + mask (immediate-offset byte loads off running pointer).
    float x[2][8];
#pragma unroll
    for (int nt = 0; nt < 2; ++nt)
#pragma unroll
      for (int r = 0; r < 8; ++r) {
        float v = sc[nt][r] * scale;
        if (mp[r * SEQ + nt * 16]) v = -1e9f;
        x[nt][r] = v;
      }
    mp += 32;

    // Online softmax; row elements live across a 16-lane half (xor<=8).
    float alpha[8];
#pragma unroll
    for (int r = 0; r < 8; ++r) {
      float mx = fmaxf(x[0][r], x[1][r]);
#pragma unroll
      for (int d = 8; d >= 1; d >>= 1) mx = fmaxf(mx, __shfl_xor(mx, d, 32));
      float nm = fmaxf(mrow[r], mx);
      alpha[r] = __expf(mrow[r] - nm);
      mrow[r] = nm;
    }
    float ps[2][8];
#pragma unroll
    for (int r = 0; r < 8; ++r) {
      float p0 = __expf(x[0][r] - mrow[r]);
      float p1 = __expf(x[1][r] - mrow[r]);
      ps[0][r] = p0; ps[1][r] = p1;
      float s = p0 + p1;
#pragma unroll
      for (int d = 8; d >= 1; d >>= 1) s += __shfl_xor(s, d, 32);
      lrow[r] = alpha[r] * lrow[r] + s;
    }
#pragma unroll
    for (int vt = 0; vt < 4; ++vt)
#pragma unroll
      for (int r = 0; r < 8; ++r) o[vt][r] *= alpha[r];

    // C-layout P -> per-wave LDS -> A-layout fragment (DS in-order in-wave).
#pragma unroll
    for (int nt = 0; nt < 2; ++nt)
#pragma unroll
      for (int r = 0; r < 8; ++r)
        Ps[wave][r + lh * 8][nt * 16 + ll] = f32_to_bf16(ps[nt][r]);
    const unsigned short* pr = &Ps[wave][ll][0];
    v16u pa = make_frag(pr + (lh << 4), pr + (lh << 4) + 8);

    // O += P (16x32) . V (32x64); kv contiguous in col-major Vst.
#pragma unroll
    for (int vt = 0; vt < 4; ++vt) {
      const unsigned short* vr = &Vst[vt * 16 + ll][lh << 4];
      o[vt] = wmma_bf16(pa, make_frag(vr, vr + 8), o[vt]);
    }
  }

  // Normalize and store ctx (bf16, [B, S, H*DV] layout).
#pragma unroll
  for (int vt = 0; vt < 4; ++vt)
#pragma unroll
    for (int r = 0; r < 8; ++r) {
      int row = q0 + r + lh * 8;
      float v = o[vt][r] / lrow[r];
      ctx[((size_t)b * SEQ + row) * rowstride + (size_t)h * DHEAD + vt * 16 + ll] =
          f32_to_bf16(v);
    }
}

// ---------------------------------------------------------------------------
// Launch
// ---------------------------------------------------------------------------
extern "C" void kernel_launch(void* const* d_in, const int* in_sizes, int n_in,
                              void* d_out, int out_size, void* d_ws, size_t ws_size,
                              hipStream_t stream) {
  const float* Q  = (const float*)d_in[0];
  const float* K  = (const float*)d_in[1];
  const float* V  = (const float*)d_in[2];
  const unsigned char* mask = (const unsigned char*)d_in[3];  // bool tensor
  const float* Wq = (const float*)d_in[4];
  const float* bq = (const float*)d_in[5];
  const float* Wk = (const float*)d_in[6];
  const float* bk = (const float*)d_in[7];
  const float* Wv = (const float*)d_in[8];
  const float* bv = (const float*)d_in[9];
  const float* Wo = (const float*)d_in[10];
  const float* bo = (const float*)d_in[11];
  float* out = (float*)d_out;

  // Workspace layout (ushort elements).
  unsigned short* w = (unsigned short*)d_ws;
  const size_t NQKV = (size_t)MROWS * DMODEL;      // 4M elements
  const size_t NW   = (size_t)DMODEL * DMODEL;     // 1M elements
  unsigned short* Qb   = w;
  unsigned short* Kb   = Qb + NQKV;
  unsigned short* Vb   = Kb + NQKV;
  unsigned short* Wqb  = Vb + NQKV;
  unsigned short* Wkb  = Wqb + NW;
  unsigned short* Wvb  = Wkb + NW;
  unsigned short* Wob  = Wvb + NW;
  unsigned short* qpj  = Wob + NW;
  unsigned short* kpj  = qpj + NQKV;
  unsigned short* vpj  = kpj + NQKV;
  unsigned short* ctxb = vpj + NQKV;

  // 1) fp32 -> bf16 conversions (vectorized x4).
  cvt_f32_bf16<<<1024, 256, 0, stream>>>(Q,  Qb,  (int)(NQKV / 4));
  cvt_f32_bf16<<<1024, 256, 0, stream>>>(K,  Kb,  (int)(NQKV / 4));
  cvt_f32_bf16<<<1024, 256, 0, stream>>>(V,  Vb,  (int)(NQKV / 4));
  cvt_f32_bf16<<<512,  256, 0, stream>>>(Wq, Wqb, (int)(NW / 4));
  cvt_f32_bf16<<<512,  256, 0, stream>>>(Wk, Wkb, (int)(NW / 4));
  cvt_f32_bf16<<<512,  256, 0, stream>>>(Wv, Wvb, (int)(NW / 4));
  cvt_f32_bf16<<<512,  256, 0, stream>>>(Wo, Wob, (int)(NW / 4));

  // 2) Projections (bf16 out, [B,S,H*DK] layout).
  dim3 gGemm(DMODEL / TN, MROWS / TM);
  gemm_bf16<<<gGemm, 256, 0, stream>>>(Qb, Wqb, bq, qpj, nullptr,
                                       MROWS, DMODEL, DMODEL, 0);
  gemm_bf16<<<gGemm, 256, 0, stream>>>(Kb, Wkb, bk, kpj, nullptr,
                                       MROWS, DMODEL, DMODEL, 0);
  gemm_bf16<<<gGemm, 256, 0, stream>>>(Vb, Wvb, bv, vpj, nullptr,
                                       MROWS, DMODEL, DMODEL, 0);

  // 3) Flash attention (TDM-staged K blocks).
  dim3 gAttn(SEQ / 64, NH, BATCH);
  attn_kernel<<<gAttn, 128, 0, stream>>>(qpj, kpj, vpj, mask, ctxb);

  // 4) Output projection (f32 out).
  gemm_bf16<<<gGemm, 256, 0, stream>>>(ctxb, Wob, bo, nullptr, out,
                                       MROWS, DMODEL, DMODEL, 1);
}
